// DSC_17532056502657
// MI455X (gfx1250) — matrix operand order repaired and checked by reading, not compiled, not run on hardware
//
#include <hip/hip_runtime.h>

typedef unsigned short u16t;
typedef __attribute__((ext_vector_type(16))) __bf16 v16bf;
typedef __attribute__((ext_vector_type(8)))  float  v8f;
typedef __attribute__((ext_vector_type(4)))  unsigned int v4u;
typedef __attribute__((ext_vector_type(8)))  int v8i_t;
typedef __attribute__((ext_vector_type(4)))  int v4i_t;

#define DD    512   // state dim
#define PP    256   // obs dim
#define MCC   128   // control dim
#define HH    32
#define MMH   64
#define TSTEPS 1024
#define HP1   33
#define NWG   8

// ---------- bf16 helpers ----------
__device__ __forceinline__ u16t f2bf(float f) {
  unsigned u = __float_as_uint(f);
  unsigned r = u + 0x7FFFu + ((u >> 16) & 1u);   // RNE
  return (u16t)(r >> 16);
}
__device__ __forceinline__ float bf2f(u16t h) {
  return __uint_as_float(((unsigned)h) << 16);
}

union VB { v16bf v; u16t s[16]; uint4 q[2]; };

// ---------- WMMA operand loaders (bf16 16x16x32, A/B k order is contiguous k = 16h + s, not the ISA table order) ----------
// A 16x32: lanes 0-15 -> M=lane, K=k0..k0+15; lanes 16-31 -> K=k0+16..k0+31
__device__ __forceinline__ v16bf load_a_tile(const u16t* base, int stride, int m0, int k0, int lane) {
  VB u;
  const u16t* p = base + (size_t)(m0 + (lane & 15)) * stride + k0 + ((lane & 16) ? 16 : 0);
  u.q[0] = *(const uint4*)(p);
  u.q[1] = *(const uint4*)(p + 8);
  return u.v;
}
// B 32x16 row-major [k][n]: lane owns column n0+(lane&15); lanes 0-15 K=k0..k0+15, lanes 16-31 K=k0+16..k0+31
__device__ __forceinline__ v16bf load_b_tile(const u16t* base, int stride, int k0, int n0, int lane) {
  VB u;
  int n = n0 + (lane & 15);
  const u16t* p = base + (size_t)(k0 + ((lane & 16) ? 16 : 0)) * stride + n;
#pragma unroll
  for (int v = 0; v < 16; ++v) u.s[v] = p[(size_t)v * stride];
  return u.v;
}
// D 16x16 f32: lane -> N, VGPR r -> M = m0 + r (lanes 0-15) / m0+8+r (lanes 16-31)
__device__ __forceinline__ void store_d_bf16(u16t* out, int stride, int m0, int n0, int lane, v8f acc) {
  int n  = n0 + (lane & 15);
  int mb = m0 + ((lane & 16) ? 8 : 0);
#pragma unroll
  for (int r = 0; r < 8; ++r) out[(size_t)(mb + r) * stride + n] = f2bf(acc[r]);
}

// ---------- TDM: 1-D dword tile copy global -> LDS (D# per CDNA5 ISA ch.8) ----------
// group0: [1:0]=count=1, [63:32]=lds_addr, [120:64]=global_addr, [127:126]=type=2
// group1: [17:16]=data_size(2 -> 4B), [79:48]=tensor_dim0, [111:80]=tensor_dim1,
//         [127:112]=tile_dim0, [143:128]=tile_dim1, [207:160]=tensor_dim0_stride
__device__ __forceinline__ void tdm_copy_1d_dwords(void* lds_dst, const void* gsrc, unsigned ndw) {
  unsigned lds_addr = (unsigned)(size_t)lds_dst;           // low 32b = LDS byte offset
  unsigned long long ga = (unsigned long long)(size_t)gsrc;
  v4u g0;
  g0.x = 1u;                                               // count=1, user descriptor
  g0.y = lds_addr;
  g0.z = (unsigned)(ga & 0xFFFFFFFFull);
  g0.w = (unsigned)((ga >> 32) & 0x01FFFFFFull) | (2u << 30); // type=2 ("image")
  v8i_t g1;
  g1[0] = (int)(2u << 16);                                 // data_size=4B, wg_mask=0, no flags
  g1[1] = (int)((ndw & 0xFFFFu) << 16);                    // tensor_dim0[15:0]
  g1[2] = (int)(((ndw >> 16) & 0xFFFFu) | (1u << 16));     // tensor_dim0[31:16] | tensor_dim1=1
  g1[3] = (int)((ndw & 0xFFFFu) << 16);                    // tensor_dim1 hi=0 | tile_dim0=ndw
  g1[4] = 1;                                               // tile_dim1=1, tile_dim2=0
  g1[5] = (int)ndw;                                        // tensor_dim0_stride[31:0]
  g1[6] = 0;
  g1[7] = 0;
  v4i_t z4 = {0, 0, 0, 0};
  v8i_t z8 = {0, 0, 0, 0, 0, 0, 0, 0};
  __builtin_amdgcn_tensor_load_to_lds(g0, g1, z4, z4, z8, 0);  // 6-arg form (clang-23)
}

// ---------- wave sub-reductions ----------
__device__ __forceinline__ float red8(float v)  { v += __shfl_xor(v,4,8);  v += __shfl_xor(v,2,8);  v += __shfl_xor(v,1,8);  return v; }
__device__ __forceinline__ float red16(float v) { v += __shfl_xor(v,8,16); v += __shfl_xor(v,4,16); v += __shfl_xor(v,2,16); v += __shfl_xor(v,1,16); return v; }
__device__ __forceinline__ float red4(float v)  { v += __shfl_xor(v,2,4);  v += __shfl_xor(v,1,4);  return v; }

// ---------- grid barrier: monotonic counter, agent-scope acq/rel ----------
__device__ __forceinline__ void gbar(unsigned* bar, unsigned target) {
  __syncthreads();
  if (threadIdx.x == 0) {
    __hip_atomic_fetch_add(bar, 1u, __ATOMIC_ACQ_REL, __HIP_MEMORY_SCOPE_AGENT);
    while (__hip_atomic_load(bar, __ATOMIC_ACQUIRE, __HIP_MEMORY_SCOPE_AGENT) < target) {
      __builtin_amdgcn_s_sleep(2);
    }
  }
  __syncthreads();
}

// =====================================================================
// prep: bf16 conversions, M2 reduction, state init
// =====================================================================
__global__ __launch_bounds__(256) void prep_kernel(
    const float* A, const float* Bm, const float* Cm, const float* Sg,
    const float* sm, const float* x0, const float* Mt,
    u16t* Abf, u16t* Cbf, u16t* AB0, u16t* sigbf, u16t* M2bf,
    float* xbuf, float* u_ring, u16t* y_ring, u16t* y_proj,
    float* y_obs, float* u_vec, unsigned* bar)
{
  int gt = blockIdx.x * 256 + threadIdx.x;
  int gs = gridDim.x * 256;
  for (int i = gt; i < DD*DD;  i += gs) Abf[i]   = f2bf(A[i]);
  for (int i = gt; i < PP*DD;  i += gs) Cbf[i]   = f2bf(Cm[i]);
  for (int i = gt; i < DD*MCC; i += gs) AB0[i]   = f2bf(Bm[i]);
  for (int i = gt; i < HH*MMH; i += gs) sigbf[i] = f2bf(Sg[i]);
  // M2[c,h,q] = sum_j Mt[c,h,j,q] * sm[j]   (coalesced over q)
  for (int i = gt; i < MCC*HH*PP; i += gs) {
    int q = i & (PP-1);
    int r = i >> 8;                    // r = c*HH + h
    const float* base = Mt + (size_t)r * (HH*PP) + q;
    float acc = 0.f;
#pragma unroll 8
    for (int j = 0; j < HH; ++j) acc += base[(size_t)j * PP] * sm[j];
    M2bf[i] = f2bf(acc);
  }
  for (int i = gt; i < DD; i += gs) { xbuf[i] = x0[i]; xbuf[DD + i] = 0.f; }
  for (int i = gt; i < HP1*MCC; i += gs) u_ring[i] = 0.f;
  for (int i = gt; i < MMH*PP;  i += gs) y_ring[i] = 0;
  for (int i = gt; i < HH*PP;   i += gs) y_proj[i] = 0;
  for (int i = gt; i < PP;      i += gs) y_obs[i]  = 0.f;
  for (int i = gt; i < MCC;     i += gs) u_vec[i]  = 0.f;
  if (gt == 0) *bar = 0u;
}

// =====================================================================
// CAB precompute step i:  CAB_i = C @ AB,  AB' = A @ AB   (bf16 WMMA, f32 accum)
// 384 tile-waves: 128 for CAB (16x8 tiles), 256 for AB' (32x8 tiles), K=512
// =====================================================================
__global__ __launch_bounds__(256) void cab_kernel(
    const u16t* Abf, const u16t* Cbf, const u16t* ABcur,
    u16t* ABnext, u16t* CABi, int do_next)
{
  int lane = threadIdx.x & 31;
  int w = blockIdx.x * 8 + (threadIdx.x >> 5);
  if (w < 128) {
    int m0 = (w >> 3) * 16, n0 = (w & 7) * 16;
    v8f acc = {};
    for (int k0 = 0; k0 < DD; k0 += 32) {
      if (k0 + 32 < DD)
        __builtin_prefetch(Cbf + (size_t)(m0 + (lane & 15)) * DD + k0 + 32, 0, 1);
      v16bf a = load_a_tile(Cbf, DD, m0, k0, lane);
      v16bf b = load_b_tile(ABcur, MCC, k0, n0, lane);
      acc = __builtin_amdgcn_wmma_f32_16x16x32_bf16(false, a, false, b, (short)0, acc, false, false);
    }
    store_d_bf16(CABi, MCC, m0, n0, lane, acc);
  } else if (do_next) {
    int w2 = w - 128;
    int m0 = (w2 >> 3) * 16, n0 = (w2 & 7) * 16;
    v8f acc = {};
    for (int k0 = 0; k0 < DD; k0 += 32) {
      if (k0 + 32 < DD)
        __builtin_prefetch(Abf + (size_t)(m0 + (lane & 15)) * DD + k0 + 32, 0, 1);
      v16bf a = load_a_tile(Abf, DD, m0, k0, lane);
      v16bf b = load_b_tile(ABcur, MCC, k0, n0, lane);
      acc = __builtin_amdgcn_wmma_f32_16x16x32_bf16(false, a, false, b, (short)0, acc, false, false);
    }
    store_d_bf16(ABnext, MCC, m0, n0, lane, acc);
  }
}

// =====================================================================
// persistent sequential simulator: NWG blocks x 256 threads, T steps,
// 4 grid barriers per step; TDM stages the big LDS tiles
// =====================================================================
__global__ __launch_bounds__(256) void sim_kernel(
    const float* Amat, const float* Bmat, const float* Cmat,
    const float* Qmat, const float* Rmat, const float* Kmat,
    const u16t* CAB, const u16t* M2, const u16t* sigbf,
    float* xbuf, float* u_ring, u16t* y_ring, u16t* y_proj,
    float* y_obs, float* u_vec, unsigned* bar, float* out)
{
  __shared__ __attribute__((aligned(16))) char smem[19456];
  const int tid  = threadIdx.x;
  const int wg   = blockIdx.x;
  const int gid  = wg * 256 + tid;
  const int lane = tid & 31;
  const int wv   = gid >> 5;              // global wave id 0..63
  unsigned epoch = 0;

  for (int t = 0; t < TSTEPS; ++t) {
    const float* xc = xbuf + (t & 1) * DD;
    float*       xn = xbuf + ((t + 1) & 1) * DD;

    // ---------- Phase A: y_obs[p] = C·x ; y_nat[p] = y_obs - CAB⊙u_ring ----------
    {
      float* sx = (float*)smem;           // 512 f32
      float* su = sx + DD;                // 33*128 f32
      if (tid < 32) tdm_copy_1d_dwords(su, u_ring, HP1 * MCC);  // async DMA to LDS
      for (int i = tid; i < DD; i += 256) sx[i] = xc[i];
      if (tid < 32) __builtin_amdgcn_s_wait_tensorcnt(0);
      __syncthreads();

      int p = wg * 32 + (tid >> 3);
      int s = tid & 7;
      float ayo = 0.f, acv = 0.f;
      {
        const float4* Cr = (const float4*)(Cmat + (size_t)p * DD + s * 64);
        const float4* Xr = (const float4*)(sx + s * 64);
#pragma unroll 4
        for (int k = 0; k < 16; ++k) {
          float4 c4 = Cr[k], x4 = Xr[k];
          ayo += c4.x * x4.x + c4.y * x4.y + c4.z * x4.z + c4.w * x4.w;
        }
      }
      for (int i = s; i < HP1; i += 8) {
        if (i + 8 < HP1)
          __builtin_prefetch(CAB + ((size_t)(i + 8) * PP + p) * MCC, 0, 1);
        int slot = (t - 1 - i + 66) % 33;
        const uint4* cq = (const uint4*)(CAB + ((size_t)i * PP + p) * MCC);
        const float* ur = su + slot * MCC;
#pragma unroll 4
        for (int c8 = 0; c8 < 16; ++c8) {
          uint4 wq = cq[c8];
          const float* u8 = ur + c8 * 8;
          acv += bf2f((u16t)(wq.x & 0xFFFF)) * u8[0] + bf2f((u16t)(wq.x >> 16)) * u8[1];
          acv += bf2f((u16t)(wq.y & 0xFFFF)) * u8[2] + bf2f((u16t)(wq.y >> 16)) * u8[3];
          acv += bf2f((u16t)(wq.z & 0xFFFF)) * u8[4] + bf2f((u16t)(wq.z >> 16)) * u8[5];
          acv += bf2f((u16t)(wq.w & 0xFFFF)) * u8[6] + bf2f((u16t)(wq.w >> 16)) * u8[7];
        }
      }
      ayo = red8(ayo);
      acv = red8(acv);
      if (s == 0) {
        y_obs[p] = ayo;
        y_ring[(size_t)(t & 63) * PP + p] = f2bf(ayo - acv);   // prepend y_nat_t
      }
    }
    gbar(bar, NWG * ++epoch);             // barrier 1

    // ---------- Phase B (WMMA): y_proj(32x256) = sigma_phi(32x64) @ y_buf(64x256) ----------
    if (wv < 32) {
      int h0 = (wv >> 4) * 16;
      int q0 = (wv & 15) * 16;
      v8f acc = {};
#pragma unroll
      for (int k0 = 0; k0 < MMH; k0 += 32) {
        v16bf a = load_a_tile(sigbf, MMH, h0, k0, lane);
        VB ub;                             // B[k=m][n=q], ring-rotated gather
        int n  = q0 + (lane & 15);
        int kb = k0 + ((lane & 16) ? 16 : 0);
#pragma unroll
        for (int v = 0; v < 16; ++v) {
          int m = kb + v;
          int slot = (t - m + 64) & 63;
          ub.s[v] = y_ring[(size_t)slot * PP + n];
        }
        acc = __builtin_amdgcn_wmma_f32_16x16x32_bf16(false, a, false, ub.v, (short)0, acc, false, false);
      }
      store_d_bf16(y_proj, PP, h0, q0, lane, acc);
    }
    gbar(bar, NWG * ++epoch);             // barrier 2

    // ---------- Phase C: u[c] = M2[c]⊙y_proj − K[c]·y_obs ----------
    {
      u16t*  spj = (u16t*)smem;                     // 8192 bf16
      float* syo = (float*)(smem + 16384);          // 256 f32
      if (tid < 32) tdm_copy_1d_dwords(spj, y_proj, (HH * PP) / 2);  // 4096 dwords
      for (int i = tid; i < PP; i += 256) syo[i] = y_obs[i];
      if (tid < 32) __builtin_amdgcn_s_wait_tensorcnt(0);
      __syncthreads();

      int c = gid >> 4;
      int s = gid & 15;
      float au = 0.f;
      {
        const uint4* m2q = (const uint4*)(M2 + (size_t)c * (HH * PP) + s * 512);
        const uint4* pjq = (const uint4*)(spj + s * 512);
#pragma unroll 4
        for (int k8 = 0; k8 < 64; ++k8) {
          uint4 wq = m2q[k8], xq = pjq[k8];
          au += bf2f((u16t)(wq.x & 0xFFFF)) * bf2f((u16t)(xq.x & 0xFFFF)) + bf2f((u16t)(wq.x >> 16)) * bf2f((u16t)(xq.x >> 16));
          au += bf2f((u16t)(wq.y & 0xFFFF)) * bf2f((u16t)(xq.y & 0xFFFF)) + bf2f((u16t)(wq.y >> 16)) * bf2f((u16t)(xq.y >> 16));
          au += bf2f((u16t)(wq.z & 0xFFFF)) * bf2f((u16t)(xq.z & 0xFFFF)) + bf2f((u16t)(wq.z >> 16)) * bf2f((u16t)(xq.z >> 16));
          au += bf2f((u16t)(wq.w & 0xFFFF)) * bf2f((u16t)(xq.w & 0xFFFF)) + bf2f((u16t)(wq.w >> 16)) * bf2f((u16t)(xq.w >> 16));
        }
      }
      float ak = 0.f;
      {
        const float* Kr = Kmat + (size_t)c * PP + s * 16;
        const float* yr = syo + s * 16;
#pragma unroll
        for (int k = 0; k < 16; ++k) ak += Kr[k] * yr[k];
      }
      au = red16(au);
      ak = red16(ak);
      if (s == 0) {
        float uc = au - ak;
        u_vec[c] = uc;
        u_ring[(size_t)(t % 33) * MCC + c] = uc;    // prepend u_t
      }
    }
    gbar(bar, NWG * ++epoch);             // barrier 3

    // ---------- Phase D: x' = A·x + B·u ; cost (WG0) ----------
    {
      float* sx2  = (float*)smem;                   // 512
      float* suv  = (float*)(smem + 2048);          // 128
      float* syo2 = (float*)(smem + 2560);          // 256
      float* cred = (float*)(smem + 3584);          // 256
      for (int i = tid; i < DD;  i += 256) sx2[i] = xc[i];
      for (int i = tid; i < MCC; i += 256) suv[i] = u_vec[i];
      for (int i = tid; i < PP;  i += 256) syo2[i] = y_obs[i];
      __syncthreads();

      int d = gid >> 2;
      int s = gid & 3;
      float ax = 0.f;
      {
        const float4* Ar = (const float4*)(Amat + (size_t)d * DD + s * 128);
        const float4* Xr = (const float4*)(sx2 + s * 128);
#pragma unroll 4
        for (int k = 0; k < 32; ++k) {
          float4 a4 = Ar[k], x4 = Xr[k];
          ax += a4.x * x4.x + a4.y * x4.y + a4.z * x4.z + a4.w * x4.w;
        }
        const float4* Br = (const float4*)(Bmat + (size_t)d * MCC + s * 32);
        const float4* Ur = (const float4*)(suv + s * 32);
#pragma unroll
        for (int k = 0; k < 8; ++k) {
          float4 b4 = Br[k], u4 = Ur[k];
          ax += b4.x * u4.x + b4.y * u4.y + b4.z * u4.z + b4.w * u4.w;
        }
      }
      ax = red4(ax);
      if (s == 0) xn[d] = ax;

      if (wg == 0) {                       // cost_t = y'Qy + u'Ru
        int p = tid;
        float qy = 0.f;
        const float4* Qr = (const float4*)(Qmat + (size_t)p * PP);
#pragma unroll 4
        for (int k = 0; k < 64; ++k) {
          float4 q4 = Qr[k];
          const float* y4 = syo2 + k * 4;
          qy += q4.x * y4[0] + q4.y * y4[1] + q4.z * y4[2] + q4.w * y4[3];
        }
        float contrib = syo2[p] * qy;
        if (p < MCC) {
          float ru = 0.f;
          const float4* Rr = (const float4*)(Rmat + (size_t)p * MCC);
#pragma unroll 4
          for (int k = 0; k < 32; ++k) {
            float4 r4 = Rr[k];
            const float* u4 = suv + k * 4;
            ru += r4.x * u4[0] + r4.y * u4[1] + r4.z * u4[2] + r4.w * u4[3];
          }
          contrib += suv[p] * ru;
        }
        cred[tid] = contrib;
        __syncthreads();
        for (int st = 128; st > 0; st >>= 1) {
          if (tid < st) cred[tid] += cred[tid + st];
          __syncthreads();
        }
        if (tid == 0) out[t] = cred[0];
      }
    }
    gbar(bar, NWG * ++epoch);             // barrier 4 (x' visible for next step)
  }
}

// =====================================================================
extern "C" void kernel_launch(void* const* d_in, const int* in_sizes, int n_in,
                              void* d_out, int out_size, void* d_ws, size_t ws_size,
                              hipStream_t stream) {
  const float* A  = (const float*)d_in[0];
  const float* Bm = (const float*)d_in[1];
  const float* Cm = (const float*)d_in[2];
  const float* Qm = (const float*)d_in[3];
  const float* Rm = (const float*)d_in[4];
  const float* Km = (const float*)d_in[5];
  const float* Mt = (const float*)d_in[6];
  const float* Sg = (const float*)d_in[7];
  const float* sm = (const float*)d_in[8];
  const float* x0 = (const float*)d_in[9];
  float* out = (float*)d_out;

  char* ws = (char*)d_ws;
  size_t off = 0;
  auto alloc = [&](size_t bytes) -> char* {
    off = (off + 255) & ~(size_t)255;
    char* p = ws + off;
    off += bytes;
    return p;
  };
  u16t* Abf    = (u16t*)alloc((size_t)DD * DD * 2);
  u16t* Cbf    = (u16t*)alloc((size_t)PP * DD * 2);
  u16t* ABbuf  = (u16t*)alloc((size_t)2 * DD * MCC * 2);
  u16t* CABbf  = (u16t*)alloc((size_t)HP1 * PP * MCC * 2);
  u16t* M2bf   = (u16t*)alloc((size_t)MCC * HH * PP * 2);
  u16t* sigbf  = (u16t*)alloc((size_t)HH * MMH * 2);
  float* xbuf  = (float*)alloc((size_t)2 * DD * 4);
  float* uring = (float*)alloc((size_t)HP1 * MCC * 4);
  u16t* yring  = (u16t*)alloc((size_t)MMH * PP * 2);
  u16t* yproj  = (u16t*)alloc((size_t)HH * PP * 2);
  float* yobs  = (float*)alloc((size_t)PP * 4);
  float* uvec  = (float*)alloc((size_t)MCC * 4);
  unsigned* bar = (unsigned*)alloc(256);

  prep_kernel<<<2048, 256, 0, stream>>>(A, Bm, Cm, Sg, sm, x0, Mt,
      Abf, Cbf, ABbuf, sigbf, M2bf, xbuf, uring, yring, yproj, yobs, uvec, bar);

  for (int i = 0; i < HP1; ++i) {
    const u16t* cur = ABbuf + (size_t)(i & 1) * DD * MCC;
    u16t* nxt       = ABbuf + (size_t)((i & 1) ^ 1) * DD * MCC;
    cab_kernel<<<48, 256, 0, stream>>>(Abf, Cbf, cur, nxt,
        CABbf + (size_t)i * PP * MCC, (i < HP1 - 1) ? 1 : 0);
  }

  sim_kernel<<<NWG, 256, 0, stream>>>(A, Bm, Cm, Qm, Rm, Km,
      CABbf, M2bf, sigbf, xbuf, uring, yring, yproj, yobs, uvec, bar, out);
}
